// TransformerEncoderLayer_2027224564616
// MI455X (gfx1250) — matrix operand order repaired and checked by reading, not compiled, not run on hardware
//
#include <hip/hip_runtime.h>
#include <hip/hip_bf16.h>
#include <math.h>

typedef __bf16 bf16;
typedef __attribute__((ext_vector_type(16))) __bf16   v16bf;
typedef __attribute__((ext_vector_type(8)))  float    v8f;
typedef __attribute__((ext_vector_type(8)))  unsigned v8u;
typedef __attribute__((ext_vector_type(4)))  unsigned tdm_v4u;
typedef __attribute__((ext_vector_type(8)))  int      tdm_v8i;
typedef __attribute__((ext_vector_type(4)))  int      tdm_v4i;

#define DIM_   1024
#define HEADS_ 16
#define HD_    64
#define HID_   4096
#define B_     2
#define T_     2048
#define M_     (B_ * T_)

#if __has_builtin(__builtin_amdgcn_tensor_load_to_lds)
#define HAVE_TDM 1
#else
#define HAVE_TDM 0
#endif

// ---------------------------------------------------------------------------
// CDNA5 data-movement helpers (guarded: degrade gracefully if builtin absent)
// ---------------------------------------------------------------------------
__device__ __forceinline__ void cp_async16(const bf16* g, bf16* l)
{
#if __has_builtin(__builtin_amdgcn_global_load_async_to_lds_b128)
    __builtin_amdgcn_global_load_async_to_lds_b128(
        (__attribute__((address_space(1))) tdm_v4i*)g,
        (__attribute__((address_space(3))) tdm_v4i*)l, 0, 0);
#else
    *(uint4*)l = *(const uint4*)g;
#endif
}

__device__ __forceinline__ void wait_async()
{
#if __has_builtin(__builtin_amdgcn_s_wait_asynccnt)
    __builtin_amdgcn_s_wait_asynccnt(0);
#else
    asm volatile("s_wait_asynccnt 0x0" ::: "memory");
#endif
}

__device__ __forceinline__ unsigned lds_offset(const void* p)
{
    return (unsigned)(unsigned long long)(__attribute__((address_space(3))) void*)p;
}

#if HAVE_TDM
// 2D tile load via Tensor Data Mover. Loads tile1 rows x tile0 elements (bf16)
// from a row-major tensor with stride0 elements/row into LDS at ldsOff, adding
// (padAmt+1) DWORDs of LDS padding every (1<<(padInt+1)) DWORDs stored.
__device__ __forceinline__ void tdm_load_2d(const bf16* gsrc, unsigned ldsOff,
                                            unsigned tile0, unsigned tile1,
                                            unsigned long long stride0,
                                            unsigned padInt, unsigned padAmt)
{
    unsigned long long ga = (unsigned long long)gsrc;
    unsigned long long td0 = stride0;        // tensor_dim0 (>= tile0, no OOB)
    unsigned long long td1 = 1u << 20;       // tensor_dim1 (large, no OOB)
    tdm_v4u g0;
    g0[0] = 1u;                                               // count=1, user mode
    g0[1] = ldsOff;                                           // lds_addr
    g0[2] = (unsigned)(ga & 0xffffffffu);                     // global_addr lo
    g0[3] = (unsigned)((ga >> 32) & 0x01ffffffu) | (2u << 30);// global_addr hi | type=2
    tdm_v8i g1;
    g1[0] = (int)((1u << 16)                // data_size = 2 bytes
                | (1u << 20)                // pad_enable
                | (padInt << 22)            // pad_interval code
                | (padAmt << 25));          // pad_amount code
    g1[1] = (int)((td0 & 0xffffu) << 16);                       // tdim0[15:0] @63:48
    g1[2] = (int)(((td0 >> 16) & 0xffffu) | ((td1 & 0xffffu) << 16));
    g1[3] = (int)(((td1 >> 16) & 0xffffu) | (tile0 << 16));     // tile_dim0 @127:112
    g1[4] = (int)(tile1 & 0xffffu);                             // tile_dim1 (tile_dim2=0)
    g1[5] = (int)(stride0 & 0xffffffffu);                       // tdim0_stride lo
    g1[6] = (int)((stride0 >> 32) & 0xffffu);                   // tdim0_stride hi
    g1[7] = 0;
    tdm_v4i z4 = {0, 0, 0, 0};
    tdm_v8i z8 = {0, 0, 0, 0, 0, 0, 0, 0};
    __builtin_amdgcn_tensor_load_to_lds(g0, g1, z4, z4, z8, 0);
}
#endif

// ---------------------------------------------------------------------------
// LDS -> WMMA fragment loaders (documented CDNA5 VGPR layouts)
// ---------------------------------------------------------------------------
__device__ __forceinline__ v16bf frag_a(const bf16* base, int row, int ldstr, int half)
{
    v8u u;
#pragma unroll
    for (int r = 0; r < 8; ++r) {
        int kb = ((r < 4) ? 0 : 16) + (r & 3) * 2 + half * 8;   // interleaved A layout
        u[r] = *(const unsigned*)(base + row * ldstr + kb);
    }
    return __builtin_bit_cast(v16bf, u);
}

__device__ __forceinline__ v16bf frag_b(const bf16* base, int row, int ldstr, int half)
{
    v8u u;
#pragma unroll
    for (int r = 0; r < 8; ++r) {
        int kb = ((r < 4) ? 0 : 16) + (r & 3) * 2 + half * 8;   // interleaved K layout (matches A)
        u[r] = *(const unsigned*)(base + row * ldstr + kb);
    }
    return __builtin_bit_cast(v16bf, u);
}

// ---------------------------------------------------------------------------
// f32 [K][N] -> bf16 [N][K] tiled transpose (weights, one-time)
// ---------------------------------------------------------------------------
__global__ __launch_bounds__(256)
void cvt_transpose_kernel(const float* __restrict__ in, bf16* __restrict__ out,
                          int Kd, int Nd)
{
    __shared__ float tile[32][33];
    int k0 = blockIdx.y * 32;
    int n0 = blockIdx.x * 32;
    int tx = threadIdx.x & 31;
    int ty = threadIdx.x >> 5;
#pragma unroll
    for (int i = 0; i < 4; ++i) {
        int k = ty + i * 8;
        tile[k][tx] = in[(size_t)(k0 + k) * Nd + n0 + tx];
    }
    __syncthreads();
#pragma unroll
    for (int i = 0; i < 4; ++i) {
        int n = ty + i * 8;
        out[(size_t)(n0 + n) * Kd + k0 + tx] = (bf16)tile[tx][n];
    }
}

// ---------------------------------------------------------------------------
// LayerNorm over last dim (1024), one block per row, bf16 output
// ---------------------------------------------------------------------------
__global__ __launch_bounds__(256)
void layernorm_kernel(const float* __restrict__ x, const float* __restrict__ w,
                      const float* __restrict__ b, bf16* __restrict__ out)
{
    constexpr int D = DIM_;
    __shared__ float rs[256];
    __shared__ float rq[256];
    int row = blockIdx.x;
    int tid = threadIdx.x;
    const float* xr = x + (size_t)row * D;
    float v[4];
    float s = 0.f, q = 0.f;
#pragma unroll
    for (int i = 0; i < 4; ++i) {
        v[i] = xr[tid + i * 256];
        s += v[i];
        q += v[i] * v[i];
    }
    rs[tid] = s; rq[tid] = q;
    __syncthreads();
    for (int off = 128; off > 0; off >>= 1) {
        if (tid < off) { rs[tid] += rs[tid + off]; rq[tid] += rq[tid + off]; }
        __syncthreads();
    }
    float mu   = rs[0] * (1.0f / D);
    float var  = rq[0] * (1.0f / D) - mu * mu;
    float rstd = rsqrtf(var + 1e-5f);
#pragma unroll
    for (int i = 0; i < 4; ++i) {
        int c = tid + i * 256;
        out[(size_t)row * D + c] = (bf16)((v[i] - mu) * rstd * w[c] + b[c]);
    }
}

// ---------------------------------------------------------------------------
// bf16 WMMA GEMM: C[M,N] = A[M,K] * Bt[N,K]^T  (both bf16, K contiguous)
// Block: 256 thr = 8 waves (4m x 2n); block tile 128x128; wave tile 32x64.
// Double-buffered LDS; A staged via async-to-LDS, B staged via TDM.
// MODE 0: store bf16 raw            (qkv)
// MODE 1: +bias +resid  -> f32      (proj residual)
// MODE 2: +bias, exact GELU -> bf16 (fc1)
// MODE 3: +bias +resid  -> f32      (fc2 -> output)
// ---------------------------------------------------------------------------
#define BLK 128
#define GBK 32
#define LDSTR 40   // padded K-stride (bf16 elems): 32 data + 8 pad

template<int MODE>
__global__ __launch_bounds__(256)
void gemm_bf16_kernel(const bf16* __restrict__ A, const bf16* __restrict__ Bt,
                      const float* __restrict__ bias, const float* __restrict__ resid,
                      float* __restrict__ outF, bf16* __restrict__ outB,
                      int M, int N, int K)
{
    __shared__ bf16 As[2][BLK * LDSTR];
    __shared__ bf16 Bs[2][BLK * LDSTR];

    int tid  = threadIdx.x;
    int lane = tid & 31;
    int wave = tid >> 5;
    int wm   = wave >> 1;      // 0..3
    int wn   = wave & 1;       // 0..1
    int half = lane >> 4;
    int lm   = lane & 15;
    int m0 = blockIdx.y * BLK;
    int n0 = blockIdx.x * BLK;

    v8f acc[2][4] = {};

    auto stage = [&](int k0, int buf) {
        int row = tid >> 1;
        int c16 = (tid & 1) * 16;
        const bf16* ga = A + (size_t)(m0 + row) * K + k0 + c16;
        bf16* la = &As[buf][row * LDSTR + c16];
        cp_async16(ga, la);
        cp_async16(ga + 8, la + 8);
#if HAVE_TDM
        if (wave == 0)  // one TDM descriptor covers the whole 128x32 B tile
            tdm_load_2d(Bt + (size_t)n0 * K + k0, lds_offset(&Bs[buf][0]),
                        GBK, BLK, (unsigned long long)K, /*padInt16DW*/3, /*padAmt4DW*/3);
#else
        const bf16* gb = Bt + (size_t)(n0 + row) * K + k0 + c16;
        bf16* lb = &Bs[buf][row * LDSTR + c16];
        cp_async16(gb, lb);
        cp_async16(gb + 8, lb + 8);
#endif
    };
    auto flush = [&]() {
        wait_async();
#if HAVE_TDM
        if (wave == 0) __builtin_amdgcn_s_wait_tensorcnt(0);
#endif
        __syncthreads();
    };

    stage(0, 0);
    flush();
    int buf = 0;
    for (int k0 = 0; k0 < K; k0 += GBK, buf ^= 1) {
        if (k0 + GBK < K) stage(k0 + GBK, buf ^ 1);   // prefetch overlaps compute

        v16bf a[2], b[4];
#pragma unroll
        for (int i = 0; i < 2; ++i)
            a[i] = frag_a(&As[buf][0], wm * 32 + i * 16 + lm, LDSTR, half);
#pragma unroll
        for (int j = 0; j < 4; ++j)
            b[j] = frag_b(&Bs[buf][0], wn * 64 + j * 16 + lm, LDSTR, half);
#pragma unroll
        for (int i = 0; i < 2; ++i)
#pragma unroll
            for (int j = 0; j < 4; ++j)
                acc[i][j] = __builtin_amdgcn_wmma_f32_16x16x32_bf16(
                    false, a[i], false, b[j], (short)0, acc[i][j], false, false);

        flush();
    }

    // epilogue: C layout -> element r: row = r + 8*half, col = lane%16
#pragma unroll
    for (int j = 0; j < 4; ++j) {
        int gn = n0 + wn * 64 + j * 16 + lm;
        float bs = (MODE != 0) ? bias[gn] : 0.f;
#pragma unroll
        for (int i = 0; i < 2; ++i)
#pragma unroll
            for (int r = 0; r < 8; ++r) {
                int gm = m0 + wm * 32 + i * 16 + r + half * 8;
                size_t idx = (size_t)gm * N + gn;
                float v = acc[i][j][r];
                if (MODE == 0) {
                    outB[idx] = (bf16)v;
                } else if (MODE == 2) {
                    float t = v + bs;
                    outB[idx] = (bf16)(0.5f * t * (1.0f + erff(t * 0.70710678118654752f)));
                } else {
                    outF[idx] = v + bs + resid[idx];
                }
            }
    }
}

// ---------------------------------------------------------------------------
// Flash attention (causal + key-padding), bf16 WMMA, f32 online softmax.
// grid: (T/128, HEADS, B); 256 thr = 8 waves; each wave owns 16 query rows.
// ---------------------------------------------------------------------------
__global__ __launch_bounds__(256)
void flash_attn_kernel(const bf16* __restrict__ qkv, const int* __restrict__ kpm,
                       bf16* __restrict__ out)
{
    constexpr int QLD = 3 * DIM_;
    __shared__ bf16 Ks[32 * 72];       // [key][hd]   stride 72
    __shared__ bf16 Vt[64 * 36];       // [hd][key]   stride 36 (transposed)
    __shared__ bf16 Ps[8 * 16 * 36];   // per-wave P scratch [16][36]

    int tid  = threadIdx.x;
    int lane = tid & 31;
    int wave = tid >> 5;
    int half = lane >> 4;
    int lm   = lane & 15;
    int q0b  = blockIdx.x * 128;
    int h    = blockIdx.y;
    int b    = blockIdx.z;
    int qrow0 = q0b + wave * 16;
    size_t rowbase = (size_t)b * T_;

    // preload Q fragments (2 K-chunks of 32 over head dim) straight from global
    v16bf qf[2];
#pragma unroll
    for (int c = 0; c < 2; ++c)
        qf[c] = frag_a(qkv + (rowbase + qrow0) * QLD + h * HD_ + c * 32, lm, QLD, half);

    v8f o[4] = {};
    float mmax[8], lsum[8];
#pragma unroll
    for (int r = 0; r < 8; ++r) { mmax[r] = -__builtin_inff(); lsum[r] = 0.f; }

    bf16* pw = &Ps[wave * 16 * 36];
    int kvend = q0b + 128;   // causal: skip tiles beyond the q tile

    for (int kv0 = 0; kv0 < kvend; kv0 += 32) {
        __syncthreads();     // previous tile fully consumed
        {
            int key = tid >> 3;          // 0..31
            int c8  = (tid & 7) * 8;     // 0..56
            // K tile: straight copy -> async to LDS
            cp_async16(qkv + (rowbase + kv0 + key) * QLD + DIM_ + h * HD_ + c8,
                       &Ks[key * 72 + c8]);
            // V tile: transpose while staging
            uint4 vv = *(const uint4*)(qkv + (rowbase + kv0 + key) * QLD + 2 * DIM_ + h * HD_ + c8);
            bf16 tmp[8];
            *(uint4*)&tmp[0] = vv;
#pragma unroll
            for (int i = 0; i < 8; ++i) Vt[(c8 + i) * 36 + key] = tmp[i];
        }
        wait_async();
        __syncthreads();

        // S = Q K^T : 2 key-subtiles of 16, contract over head dim (2 chunks)
        v8f s[2] = {};
#pragma unroll
        for (int nt = 0; nt < 2; ++nt)
#pragma unroll
            for (int c = 0; c < 2; ++c) {
                v16bf kf = frag_b(&Ks[c * 32], nt * 16 + lm, 72, half);
                s[nt] = __builtin_amdgcn_wmma_f32_16x16x32_bf16(
                    false, qf[c], false, kf, (short)0, s[nt], false, false);
            }

        // scale + key-padding mask (always)
#pragma unroll
        for (int nt = 0; nt < 2; ++nt) {
            int key = kv0 + nt * 16 + lm;
            float pad = (kpm[rowbase + key] != 0) ? -1e9f : 0.0f;
#pragma unroll
            for (int r = 0; r < 8; ++r)
                s[nt][r] = s[nt][r] * 0.125f + pad;      // 1/sqrt(64)
        }
        // causal clamp only on tiles straddling the diagonal (uniform branch)
        if (kv0 + 31 > qrow0) {
#pragma unroll
            for (int nt = 0; nt < 2; ++nt) {
                int key = kv0 + nt * 16 + lm;
#pragma unroll
                for (int r = 0; r < 8; ++r) {
                    int q = qrow0 + r + half * 8;
                    if (key > q) s[nt][r] = -__builtin_inff();
                }
            }
        }
        float pmax[8];
#pragma unroll
        for (int r = 0; r < 8; ++r) pmax[r] = fmaxf(s[0][r], s[1][r]);
#pragma unroll
        for (int r = 0; r < 8; ++r) {
            float v = pmax[r];
            v = fmaxf(v, __shfl_xor(v, 1, 16));
            v = fmaxf(v, __shfl_xor(v, 2, 16));
            v = fmaxf(v, __shfl_xor(v, 4, 16));
            v = fmaxf(v, __shfl_xor(v, 8, 16));
            pmax[r] = v;
        }
        float alpha[8];
#pragma unroll
        for (int r = 0; r < 8; ++r) {
            float mn = fmaxf(mmax[r], pmax[r]);
            alpha[r] = __expf(mmax[r] - mn);
            mmax[r]  = mn;
        }
        float psum[8] = {};
#pragma unroll
        for (int nt = 0; nt < 2; ++nt)
#pragma unroll
            for (int r = 0; r < 8; ++r) {
                float p = __expf(s[nt][r] - mmax[r]);
                psum[r] += p;
                pw[(r + half * 8) * 36 + nt * 16 + lm] = (bf16)p;
            }
#pragma unroll
        for (int r = 0; r < 8; ++r) {
            float v = psum[r];
            v += __shfl_xor(v, 1, 16);
            v += __shfl_xor(v, 2, 16);
            v += __shfl_xor(v, 4, 16);
            v += __shfl_xor(v, 8, 16);
            lsum[r] = lsum[r] * alpha[r] + v;
        }
#pragma unroll
        for (int c = 0; c < 4; ++c)
#pragma unroll
            for (int r = 0; r < 8; ++r)
                o[c][r] *= alpha[r];

        // wave-private LDS RAW: wait for P stores before A-fragment reads
        asm volatile("s_wait_dscnt 0x0" ::: "memory");

        v16bf pf = frag_a(pw, lm, 36, half);
#pragma unroll
        for (int c = 0; c < 4; ++c) {
            v16bf vf = frag_b(Vt, c * 16 + lm, 36, half);
            o[c] = __builtin_amdgcn_wmma_f32_16x16x32_bf16(
                false, pf, false, vf, (short)0, o[c], false, false);
        }
    }

    // normalize and store (bf16, [B*T][DIM], head-major columns)
#pragma unroll
    for (int r = 0; r < 8; ++r) {
        float inv = __builtin_amdgcn_rcpf(lsum[r]);
        size_t gm = rowbase + qrow0 + r + half * 8;
#pragma unroll
        for (int c = 0; c < 4; ++c)
            out[gm * DIM_ + h * HD_ + c * 16 + lm] = (bf16)(o[c][r] * inv);
    }
}

// ---------------------------------------------------------------------------
// Host orchestration
// ---------------------------------------------------------------------------
extern "C" void kernel_launch(void* const* d_in, const int* in_sizes, int n_in,
                              void* d_out, int out_size, void* d_ws, size_t ws_size,
                              hipStream_t stream)
{
    (void)in_sizes; (void)n_in; (void)out_size; (void)ws_size;

    const float* x      = (const float*)d_in[0];
    const int*   kpm    = (const int*)  d_in[1];
    const float* ln1_w  = (const float*)d_in[2];
    const float* ln1_b  = (const float*)d_in[3];
    const float* qkv_w  = (const float*)d_in[4];
    const float* proj_w = (const float*)d_in[5];
    const float* proj_b = (const float*)d_in[6];
    const float* ln2_w  = (const float*)d_in[7];
    const float* ln2_b  = (const float*)d_in[8];
    const float* fc1_w  = (const float*)d_in[9];
    const float* fc1_b  = (const float*)d_in[10];
    const float* fc2_w  = (const float*)d_in[11];
    const float* fc2_b  = (const float*)d_in[12];
    float* out = (float*)d_out;

    char* ws = (char*)d_ws;
    size_t off = 0;
    auto alloc = [&](size_t bytes) -> void* {
        off = (off + 255) & ~(size_t)255;
        void* p = ws + off;
        off += bytes;
        return p;
    };
    bf16*  xln1   = (bf16*) alloc((size_t)M_ * DIM_ * 2);
    bf16*  qkvb   = (bf16*) alloc((size_t)M_ * 3 * DIM_ * 2);
    bf16*  attnb  = (bf16*) alloc((size_t)M_ * DIM_ * 2);
    float* x1     = (float*)alloc((size_t)M_ * DIM_ * 4);
    bf16*  hb     = (bf16*) alloc((size_t)M_ * DIM_ * 2);
    bf16*  gb     = (bf16*) alloc((size_t)M_ * HID_ * 2);
    bf16*  wqkvT  = (bf16*) alloc((size_t)DIM_ * 3 * DIM_ * 2);  // [3*DIM][DIM]
    bf16*  wprojT = (bf16*) alloc((size_t)DIM_ * DIM_ * 2);      // [DIM][DIM]
    bf16*  wfc1T  = (bf16*) alloc((size_t)DIM_ * HID_ * 2);      // [HID][DIM]
    bf16*  wfc2T  = (bf16*) alloc((size_t)HID_ * DIM_ * 2);      // [DIM][HID]

    // one-time weight transpose+convert: f32 [K][N] -> bf16 [N][K]
    cvt_transpose_kernel<<<dim3(3 * DIM_ / 32, DIM_ / 32), 256, 0, stream>>>(qkv_w,  wqkvT,  DIM_, 3 * DIM_);
    cvt_transpose_kernel<<<dim3(DIM_ / 32,     DIM_ / 32), 256, 0, stream>>>(proj_w, wprojT, DIM_, DIM_);
    cvt_transpose_kernel<<<dim3(HID_ / 32,     DIM_ / 32), 256, 0, stream>>>(fc1_w,  wfc1T,  DIM_, HID_);
    cvt_transpose_kernel<<<dim3(DIM_ / 32,     HID_ / 32), 256, 0, stream>>>(fc2_w,  wfc2T,  HID_, DIM_);

    // x -> LN1 -> bf16
    layernorm_kernel<<<M_, 256, 0, stream>>>(x, ln1_w, ln1_b, xln1);
    // qkv = ln1(x) @ qkv_w  (bf16 out)
    gemm_bf16_kernel<0><<<dim3(3 * DIM_ / BLK, M_ / BLK), 256, 0, stream>>>(
        xln1, wqkvT, nullptr, nullptr, nullptr, qkvb, M_, 3 * DIM_, DIM_);
    // flash attention
    flash_attn_kernel<<<dim3(T_ / 128, HEADS_, B_), 256, 0, stream>>>(qkvb, kpm, attnb);
    // x1 = x + attn @ proj_w + proj_b  (f32)
    gemm_bf16_kernel<1><<<dim3(DIM_ / BLK, M_ / BLK), 256, 0, stream>>>(
        attnb, wprojT, proj_b, x, x1, nullptr, M_, DIM_, DIM_);
    // h = ln2(x1) (bf16)
    layernorm_kernel<<<M_, 256, 0, stream>>>(x1, ln2_w, ln2_b, hb);
    // g = gelu(h @ fc1_w + fc1_b) (bf16)
    gemm_bf16_kernel<2><<<dim3(HID_ / BLK, M_ / BLK), 256, 0, stream>>>(
        hb, wfc1T, fc1_b, nullptr, nullptr, gb, M_, HID_, DIM_);
    // out = x1 + g @ fc2_w + fc2_b (f32)
    gemm_bf16_kernel<3><<<dim3(DIM_ / BLK, M_ / BLK), 256, 0, stream>>>(
        gb, wfc2T, fc2_b, x1, out, nullptr, M_, DIM_, HID_);
}
